// ProgressiveFocusTransformer_85203561218793
// MI455X (gfx1250) — matrix operand order repaired and checked by reading, not compiled, not run on hardware
//
#include <hip/hip_runtime.h>
#include <hip/hip_bf16.h>
#include <math.h>

#define BATCH  8
#define SEQ    1024
#define CH     768
#define NH     12
#define HDIM   64
#define NSTAGE 3
#define MLPH   3072

typedef __attribute__((ext_vector_type(16))) __bf16 v16bf;
typedef __attribute__((ext_vector_type(8)))  float  v8f;

union FragBF { v16bf v; uint4 q[2]; };

__device__ __forceinline__ v8f vzero8() {
  v8f z;
#pragma unroll
  for (int i = 0; i < 8; ++i) z[i] = 0.0f;
  return z;
}

__device__ __forceinline__ unsigned short f2bf_bits(float f) {
  union { float f; unsigned int u; } v; v.f = f;
  unsigned int u = v.u;
  u += 0x7FFFu + ((u >> 16) & 1u);   // round-to-nearest-even
  return (unsigned short)(u >> 16);
}

// A-fragment (16x32 bf16): per-lane chunks at [k .. k+7] and [k+16 .. k+23]
__device__ __forceinline__ v16bf ld_fragA(const unsigned short* p) {
  FragBF f;
  f.q[0] = *(const uint4*)(p);
  f.q[1] = *(const uint4*)(p + 16);
  return f.v;
}
// B-fragment (32x16 bf16): per-lane chunks at [k .. k+7] and [k+16 .. k+23]
__device__ __forceinline__ v16bf ld_fragB(const unsigned short* p) {
  FragBF f;
  f.q[0] = *(const uint4*)(p);
  f.q[1] = *(const uint4*)(p + 16);
  return f.v;
}

#define WMMA_BF16(a, b, c) \
  __builtin_amdgcn_wmma_f32_16x16x32_bf16(false, (a), false, (b), (short)0, (c), false, false)

// ---------------------------------------------------------------- elementwise
__global__ void f32_to_bf16_kernel(const float* __restrict__ in,
                                   unsigned short* __restrict__ out, int n) {
  int i = blockIdx.x * blockDim.x + threadIdx.x;
  if (i < n) out[i] = f2bf_bits(in[i]);
}

__global__ void copy_f32_kernel(const float* __restrict__ in,
                                float* __restrict__ out, int n) {
  int i = blockIdx.x * blockDim.x + threadIdx.x;
  if (i < n) out[i] = in[i];
}

// ---------------------------------------------------------------- layernorm
// one block per row of C=768; output bf16
__global__ void ln_kernel(const float* __restrict__ x,
                          const float* __restrict__ w,
                          const float* __restrict__ b,
                          unsigned short* __restrict__ out) {
  __shared__ float red[256];
  const int row = blockIdx.x;
  const int t = threadIdx.x;
  const float* xr = x + (size_t)row * CH;

  float s = 0.0f;
  for (int i = t; i < CH; i += 256) s += xr[i];
  red[t] = s; __syncthreads();
  for (int off = 128; off; off >>= 1) { if (t < off) red[t] += red[t + off]; __syncthreads(); }
  const float mu = red[0] / (float)CH;
  __syncthreads();

  float v = 0.0f;
  for (int i = t; i < CH; i += 256) { float d = xr[i] - mu; v += d * d; }
  red[t] = v; __syncthreads();
  for (int off = 128; off; off >>= 1) { if (t < off) red[t] += red[t + off]; __syncthreads(); }
  const float rstd = rsqrtf(red[0] / (float)CH + 1e-5f);
  __syncthreads();

  unsigned short* orow = out + (size_t)row * CH;
  for (int i = t; i < CH; i += 256)
    orow[i] = f2bf_bits((xr[i] - mu) * rstd * w[i] + b[i]);
}

// ---------------------------------------------------------------- WMMA GEMM
// out[m, o] = sum_c A[m, c] * W[o, c]   (A: MxK bf16, W: NoutxK bf16, row-major)
// One wave per 32x32 output tile; 4x v_wmma_f32_16x16x32_bf16 per K-step.
// mode 0: qkv scatter (q/k [B,H,N,64] bf16, v transposed [B,H,64,N] bf16)
// mode 1: res[row,col] += focus * (acc + bias[col])
// mode 2: hidden = bf16(gelu(acc + bias))
// mode 3: dout = res + acc + bias
__global__ void gemm_bf16_kernel(const unsigned short* __restrict__ A,
                                 const unsigned short* __restrict__ W,
                                 int M, int Nout, int K, int mode,
                                 unsigned short* __restrict__ qo,
                                 unsigned short* __restrict__ ko,
                                 unsigned short* __restrict__ vto,
                                 unsigned short* __restrict__ outb,
                                 const float* __restrict__ bias,
                                 float* __restrict__ res,
                                 float* __restrict__ dout,
                                 const float* __restrict__ focusw, int stage) {
  const int lane = threadIdx.x & 31;
  const int wv = threadIdx.x >> 5;
  const int Nt = Nout >> 5;
  const int Mt = M >> 5;
  const long tile = (long)blockIdx.x * 8 + wv;
  if (tile >= (long)Mt * Nt) return;
  const int mt = (int)(tile / Nt), nt = (int)(tile % Nt);
  const int m0 = mt << 5, n0 = nt << 5;
  const int r = lane & 15, hi = lane >> 4;

  const unsigned short* a0p = A + (size_t)(m0 + r) * K + hi * 8;
  const unsigned short* a1p = a0p + (size_t)16 * K;
  const unsigned short* b0p = W + (size_t)(n0 + r) * K + hi * 8;
  const unsigned short* b1p = b0p + (size_t)16 * K;

  v8f c00 = vzero8(), c01 = vzero8(), c10 = vzero8(), c11 = vzero8();

#pragma unroll 4
  for (int k0 = 0; k0 < K; k0 += 32) {
    v16bf a0 = ld_fragA(a0p + k0);
    v16bf a1 = ld_fragA(a1p + k0);
    v16bf b0 = ld_fragB(b0p + k0);
    v16bf b1 = ld_fragB(b1p + k0);
    c00 = WMMA_BF16(a0, b0, c00);
    c01 = WMMA_BF16(a0, b1, c01);
    c10 = WMMA_BF16(a1, b0, c10);
    c11 = WMMA_BF16(a1, b1, c11);
  }

  const float focus = (focusw != nullptr) ? focusw[stage] : 0.0f;

  auto epi = [&](const v8f& c, int dm, int dn) {
#pragma unroll
    for (int rr = 0; rr < 8; ++rr) {
      const int row = m0 + dm + rr + (hi << 3);
      const int col = n0 + dn + (lane & 15);
      const float acc = c[rr];
      if (mode == 0) {
        const int bb = row >> 10, n = row & (SEQ - 1);
        const int part = col / CH;
        const int hc = col - part * CH;
        const int hh = hc >> 6, d = hc & 63;
        const size_t bh = (size_t)(bb * NH + hh);
        const unsigned short val = f2bf_bits(acc);
        if (part == 0)      qo[(bh * SEQ + n) * HDIM + d] = val;
        else if (part == 1) ko[(bh * SEQ + n) * HDIM + d] = val;
        else                vto[(bh * HDIM + d) * SEQ + n] = val;
      } else if (mode == 1) {
        res[(size_t)row * CH + col] += focus * (acc + bias[col]);
      } else if (mode == 2) {
        const float t = acc + bias[col];
        outb[(size_t)row * MLPH + col] =
            f2bf_bits(0.5f * t * (1.0f + erff(t * 0.70710678f)));
      } else {
        dout[(size_t)row * CH + col] =
            res[(size_t)row * CH + col] + acc + bias[col];
      }
    }
  };
  epi(c00, 0, 0); epi(c01, 0, 16); epi(c10, 16, 0); epi(c11, 16, 16);
}

// ---------------------------------------------------------------- attention
// grid: B*H*(SEQ/128) blocks, 256 threads = 8 waves, one wave per 16-query tile.
// Online-softmax flash attention; scores and PV via bf16 WMMA; P relaid via LDS.
__global__ void attn_kernel(const unsigned short* __restrict__ qb,
                            const unsigned short* __restrict__ kb,
                            const unsigned short* __restrict__ vtb,
                            unsigned short* __restrict__ ob) {
  __shared__ unsigned short pls[8][16][32];  // per-wave P tile (16 q x 32 keys)
  const int lane = threadIdx.x & 31;
  const int wv = threadIdx.x >> 5;
  const int r16 = lane & 15, hi = lane >> 4;

  const int qblk = blockIdx.x % (SEQ / 128);
  const int bh = blockIdx.x / (SEQ / 128);
  const int head = bh % NH, bb = bh / NH;

  const unsigned short* qh = qb + (size_t)bh * SEQ * HDIM;   // [n][d]
  const unsigned short* kh = kb + (size_t)bh * SEQ * HDIM;   // [n][d]
  const unsigned short* vh = vtb + (size_t)bh * HDIM * SEQ;  // [d][n]
  const int m0 = qblk * 128 + wv * 16;

  // Q fragments (rows m0..m0+15, d 0..31 and 32..63) stay resident
  const unsigned short* qp = qh + (size_t)(m0 + r16) * HDIM + hi * 8;
  const v16bf qa0 = ld_fragA(qp);
  const v16bf qa1 = ld_fragA(qp + 32);

  v8f o0 = vzero8(), o1 = vzero8(), o2 = vzero8(), o3 = vzero8();
  float mrow[8], lrow[8];
#pragma unroll
  for (int r = 0; r < 8; ++r) { mrow[r] = -1e30f; lrow[r] = 0.0f; }

  for (int kt = 0; kt < SEQ; kt += 32) {
    // scores for 32 keys: S0 (keys kt..+15), S1 (keys kt+16..+31)
    const unsigned short* kp0 = kh + (size_t)(kt + r16) * HDIM + hi * 8;
    const unsigned short* kp1 = kp0 + (size_t)16 * HDIM;
    v16bf kb00 = ld_fragB(kp0);       // d 0..31
    v16bf kb01 = ld_fragB(kp0 + 32);  // d 32..63
    v16bf kb10 = ld_fragB(kp1);
    v16bf kb11 = ld_fragB(kp1 + 32);

    v8f s0 = vzero8(); s0 = WMMA_BF16(qa0, kb00, s0); s0 = WMMA_BF16(qa1, kb01, s0);
    v8f s1 = vzero8(); s1 = WMMA_BF16(qa0, kb10, s1); s1 = WMMA_BF16(qa1, kb11, s1);

    // online softmax per row (row = r + 8*hi, cols across 16 lanes of a half)
#pragma unroll
    for (int r = 0; r < 8; ++r) {
      const float sa = s0[r] * 0.125f;   // HD^-0.5
      const float sb = s1[r] * 0.125f;
      float tm = fmaxf(sa, sb);
      tm = fmaxf(tm, __shfl_xor(tm, 1));
      tm = fmaxf(tm, __shfl_xor(tm, 2));
      tm = fmaxf(tm, __shfl_xor(tm, 4));
      tm = fmaxf(tm, __shfl_xor(tm, 8));
      const float mnew = fmaxf(mrow[r], tm);
      const float alpha = __expf(mrow[r] - mnew);
      const float p0 = __expf(sa - mnew);
      const float p1 = __expf(sb - mnew);
      float ts = p0 + p1;
      ts += __shfl_xor(ts, 1);
      ts += __shfl_xor(ts, 2);
      ts += __shfl_xor(ts, 4);
      ts += __shfl_xor(ts, 8);
      lrow[r] = lrow[r] * alpha + ts;
      mrow[r] = mnew;
      o0[r] *= alpha; o1[r] *= alpha; o2[r] *= alpha; o3[r] *= alpha;
      const int prow = r + (hi << 3);
      pls[wv][prow][r16] = f2bf_bits(p0);
      pls[wv][prow][r16 + 16] = f2bf_bits(p1);
    }
    asm volatile("s_wait_dscnt 0" ::: "memory");

    // P (16x32) as A fragment from LDS
    const v16bf pa = ld_fragA(&pls[wv][r16][0] + hi * 8);

    // O += P x V, 4 output d-groups of 16; V^T gives contiguous B fragments
#pragma unroll
    for (int j = 0; j < 4; ++j) {
      const unsigned short* vp = vh + (size_t)(j * 16 + r16) * SEQ + kt + hi * 8;
      v16bf vb = ld_fragB(vp);
      if (j == 0)      o0 = WMMA_BF16(pa, vb, o0);
      else if (j == 1) o1 = WMMA_BF16(pa, vb, o1);
      else if (j == 2) o2 = WMMA_BF16(pa, vb, o2);
      else             o3 = WMMA_BF16(pa, vb, o3);
    }
    asm volatile("s_wait_dscnt 0" ::: "memory");
  }

  // normalize and store bf16 into o_buf [B,N,C] at head*64 slice
#pragma unroll
  for (int r = 0; r < 8; ++r) {
    const float inv = 1.0f / lrow[r];
    const int n = m0 + r + (hi << 3);
    const size_t base = ((size_t)bb * SEQ + n) * CH + head * HDIM + r16;
    ob[base + 0]  = f2bf_bits(o0[r] * inv);
    ob[base + 16] = f2bf_bits(o1[r] * inv);
    ob[base + 32] = f2bf_bits(o2[r] * inv);
    ob[base + 48] = f2bf_bits(o3[r] * inv);
  }
}

// ---------------------------------------------------------------- launch
extern "C" void kernel_launch(void* const* d_in, const int* in_sizes, int n_in,
                              void* d_out, int out_size, void* d_ws, size_t ws_size,
                              hipStream_t stream) {
  const float* x      = (const float*)d_in[0];
  const float* qkv_w  = (const float*)d_in[1];
  const float* proj_w = (const float*)d_in[2];
  const float* proj_b = (const float*)d_in[3];
  const float* ln1_w  = (const float*)d_in[4];
  const float* ln1_b  = (const float*)d_in[5];
  const float* ln2_w  = (const float*)d_in[6];
  const float* ln2_b  = (const float*)d_in[7];
  const float* mlp_w1 = (const float*)d_in[8];
  const float* mlp_b1 = (const float*)d_in[9];
  const float* mlp_w2 = (const float*)d_in[10];
  const float* mlp_b2 = (const float*)d_in[11];
  const float* focus  = (const float*)d_in[12];
  float* out = (float*)d_out;

  const size_t MROWS = (size_t)BATCH * SEQ;  // 8192
  char* ws = (char*)d_ws;
  size_t off = 0;
  auto walloc = [&](size_t bytes) -> void* {
    void* p = ws + off;
    off = (off + bytes + 255) & ~(size_t)255;
    return p;
  };
  unsigned short* hbuf  = (unsigned short*)walloc(MROWS * CH * 2);
  unsigned short* wqkv  = (unsigned short*)walloc((size_t)NSTAGE * 3 * CH * CH * 2);
  unsigned short* wproj = (unsigned short*)walloc((size_t)NSTAGE * CH * CH * 2);
  unsigned short* wm1   = (unsigned short*)walloc((size_t)MLPH * CH * 2);
  unsigned short* wm2   = (unsigned short*)walloc((size_t)CH * MLPH * 2);
  unsigned short* qbuf  = (unsigned short*)walloc(MROWS * CH * 2);
  unsigned short* kbuf  = (unsigned short*)walloc(MROWS * CH * 2);
  unsigned short* vtbuf = (unsigned short*)walloc(MROWS * CH * 2);
  unsigned short* obuf  = (unsigned short*)walloc(MROWS * CH * 2);
  unsigned short* hid   = (unsigned short*)walloc(MROWS * MLPH * 2);
  float* res            = (float*)walloc(MROWS * CH * 4);

  // weights -> bf16
  {
    int n = NSTAGE * 3 * CH * CH;
    f32_to_bf16_kernel<<<(n + 255) / 256, 256, 0, stream>>>(qkv_w, wqkv, n);
    n = NSTAGE * CH * CH;
    f32_to_bf16_kernel<<<(n + 255) / 256, 256, 0, stream>>>(proj_w, wproj, n);
    n = MLPH * CH;
    f32_to_bf16_kernel<<<(n + 255) / 256, 256, 0, stream>>>(mlp_w1, wm1, n);
    n = CH * MLPH;
    f32_to_bf16_kernel<<<(n + 255) / 256, 256, 0, stream>>>(mlp_w2, wm2, n);
  }

  // h = LN1(x) (bf16), res = x
  ln_kernel<<<(int)MROWS, 256, 0, stream>>>(x, ln1_w, ln1_b, hbuf);
  {
    int n = (int)(MROWS * CH);
    copy_f32_kernel<<<(n + 255) / 256, 256, 0, stream>>>(x, res, n);
  }

  for (int s = 0; s < NSTAGE; ++s) {
    {  // qkv: [8192x768] x [2304x768]^T, scatter q/k/vT
      const int blocks = (8192 / 32) * (2304 / 32) / 8;
      gemm_bf16_kernel<<<blocks, 256, 0, stream>>>(
          hbuf, wqkv + (size_t)s * 3 * CH * CH, 8192, 3 * CH, CH, 0,
          qbuf, kbuf, vtbuf, nullptr, nullptr, nullptr, nullptr, nullptr, 0);
    }
    attn_kernel<<<BATCH * NH * (SEQ / 128), 256, 0, stream>>>(qbuf, kbuf, vtbuf, obuf);
    {  // proj: res += focus[s] * (o x Wp^T + b)
      const int blocks = (8192 / 32) * (768 / 32) / 8;
      gemm_bf16_kernel<<<blocks, 256, 0, stream>>>(
          obuf, wproj + (size_t)s * CH * CH, 8192, CH, CH, 1,
          nullptr, nullptr, nullptr, nullptr, proj_b + s * CH, res, nullptr, focus, s);
    }
  }

  // h2 = LN2(res) (reuse hbuf), MLP1 + GELU, MLP2 + residual -> out
  ln_kernel<<<(int)MROWS, 256, 0, stream>>>(res, ln2_w, ln2_b, hbuf);
  {
    const int blocks = (8192 / 32) * (3072 / 32) / 8;
    gemm_bf16_kernel<<<blocks, 256, 0, stream>>>(
        hbuf, wm1, 8192, MLPH, CH, 2,
        nullptr, nullptr, nullptr, hid, mlp_b1, nullptr, nullptr, nullptr, 0);
  }
  {
    const int blocks = (8192 / 32) * (768 / 32) / 8;
    gemm_bf16_kernel<<<blocks, 256, 0, stream>>>(
        hid, wm2, 8192, CH, MLPH, 3,
        nullptr, nullptr, nullptr, nullptr, mlp_b2, res, out, nullptr, 0);
  }
}
